// GraphTransformer_13889924235784
// MI455X (gfx1250) — matrix operand order repaired and checked by reading, not compiled, not run on hardware
//
#include <hip/hip_runtime.h>
#include <hip/hip_fp16.h>

// Problem constants (match reference)
#define N_NODES 100000
#define N_EDGES 1600000
#define FDIM    128
#define NGRAPH  64
#define NCLS    10

typedef __attribute__((ext_vector_type(16))) _Float16 v16h;
typedef __attribute__((ext_vector_type(8)))  float    v8f;
typedef __attribute__((ext_vector_type(4)))  _Float16 v4h;

// ---------- helpers ----------

// Monotone order-preserving float -> uint key (for atomicMax-based segment max)
__device__ __forceinline__ unsigned int fkey(float x) {
  unsigned int u = __float_as_uint(x);
  return (u & 0x80000000u) ? ~u : (u | 0x80000000u);
}
__device__ __forceinline__ float funkey(unsigned int k) {
  unsigned int u = (k & 0x80000000u) ? (k & 0x7FFFFFFFu) : ~k;
  return __uint_as_float(u);
}

// Hardware fp32 atomic add (global_atomic_add_f32, non-returning)
__device__ __forceinline__ void atomAddF(float* p, float v) {
  unsafeAtomicAdd(p, v);
}

// ---------- small utility kernels ----------

__global__ void k_cvt_f16(const float* __restrict__ in, _Float16* __restrict__ out, int n) {
  int i = blockIdx.x * blockDim.x + threadIdx.x;
  if (i < n) out[i] = (_Float16)in[i];
}

__global__ void k_copy_f32(const float* __restrict__ in, float* __restrict__ out, int n) {
  int i = blockIdx.x * blockDim.x + threadIdx.x;
  if (i < n) out[i] = in[i];
}

__global__ void k_fill_u32(unsigned int* __restrict__ p, unsigned int v, int n) {
  int i = blockIdx.x * blockDim.x + threadIdx.x;
  if (i < n) p[i] = v;
}

// ---------- WMMA GEMM: fused 4-projection  D[N x 512] = A[N x 128] @ Wcat^T + bcat ----------
// Wcat is [512][128] f16 row-major (rows are output features, contiguous in K).
// Output columns [0,128)->Qh(f16), [128,256)->Kh(f16), [256,384)->Vh(f16), [384,512)->AGG(f32, skip-proj).
// One wave32 computes one 16x16 tile; K-loop = 4 x v_wmma_f32_16x16x32_f16.
__global__ void __launch_bounds__(256)
k_gemm_proj(const _Float16* __restrict__ Ah,
            const _Float16* __restrict__ Wh,
            const float*    __restrict__ bias,
            _Float16* __restrict__ Qh,
            _Float16* __restrict__ Kh,
            _Float16* __restrict__ Vh,
            float*    __restrict__ Sout,
            int M)
{
  const int NT = 32;                                   // 512/16 column tiles
  int wave  = blockIdx.x * (blockDim.x >> 5) + (threadIdx.x >> 5);
  int lane  = threadIdx.x & 31;
  int mTile = wave / NT;
  int nTile = wave - mTile * NT;
  if (mTile * 16 >= M) return;                         // wave-uniform guard (EXEC stays all-ones)

  int half = lane >> 4;                                // half-wave selects K sub-blocks
  const _Float16* arow = Ah + (size_t)(mTile * 16 + (lane & 15)) * FDIM;
  const _Float16* wrow = Wh + (size_t)(nTile * 16 + (lane & 15)) * FDIM;

  v8f c = {};
#pragma unroll
  for (int kk = 0; kk < FDIM; kk += 32) {
    union { v16h v; uint4 u[2]; } a, b;
    // A fragment (16x32 f16): lane = M; v0..3 = K(kk + half*8 .. +7), v4..7 = K(+16 .. +23)
    a.u[0] = *(const uint4*)(arow + kk + half * 8);
    a.u[1] = *(const uint4*)(arow + kk + half * 8 + 16);
    // B fragment (32x16 f16): lane = N; v0..3 = K(kk + half*8 .. +7), v4..7 = K(+16 .. +23)
    b.u[0] = *(const uint4*)(wrow + kk + half * 8);
    b.u[1] = *(const uint4*)(wrow + kk + half * 8 + 16);
    c = __builtin_amdgcn_wmma_f32_16x16x32_f16(false, a.v, false, b.v, (short)0, c, false, false);
  }

  int col512 = nTile * 16 + (lane & 15);
  float bn = bias[col512];
  int rowBase = mTile * 16 + half * 8;                 // C layout: VGPR j -> M = j + 8*half, N = lane&15
  int cb = col512 & 127;
  if (nTile < 24) {
    _Float16* out = (nTile < 8) ? Qh : ((nTile < 16) ? Kh : Vh);
#pragma unroll
    for (int j = 0; j < 8; j++)
      out[(size_t)(rowBase + j) * FDIM + cb] = (_Float16)(c[j] + bn);
  } else {
#pragma unroll
    for (int j = 0; j < 8; j++)
      Sout[(size_t)(rowBase + j) * FDIM + cb] = c[j] + bn;
  }
}

// ---------- WMMA GEMM: gate layer1  Out[N x 128] = relu(A @ gw1^T + gb1)  (f32 out) ----------
__global__ void __launch_bounds__(256)
k_gemm_relu(const _Float16* __restrict__ Ah,
            const _Float16* __restrict__ Wh,     // [128][128] f16
            const float*    __restrict__ bias,   // [128] f32
            float* __restrict__ Out,
            int M)
{
  const int NT = 8;
  int wave  = blockIdx.x * (blockDim.x >> 5) + (threadIdx.x >> 5);
  int lane  = threadIdx.x & 31;
  int mTile = wave / NT;
  int nTile = wave - mTile * NT;
  if (mTile * 16 >= M) return;

  int half = lane >> 4;
  const _Float16* arow = Ah + (size_t)(mTile * 16 + (lane & 15)) * FDIM;
  const _Float16* wrow = Wh + (size_t)(nTile * 16 + (lane & 15)) * FDIM;

  v8f c = {};
#pragma unroll
  for (int kk = 0; kk < FDIM; kk += 32) {
    union { v16h v; uint4 u[2]; } a, b;
    a.u[0] = *(const uint4*)(arow + kk + half * 8);
    a.u[1] = *(const uint4*)(arow + kk + half * 8 + 16);
    b.u[0] = *(const uint4*)(wrow + kk + half * 8);
    b.u[1] = *(const uint4*)(wrow + kk + half * 8 + 16);
    c = __builtin_amdgcn_wmma_f32_16x16x32_f16(false, a.v, false, b.v, (short)0, c, false, false);
  }

  int col = nTile * 16 + (lane & 15);
  float bn = bias[col];
  int rowBase = mTile * 16 + half * 8;
#pragma unroll
  for (int j = 0; j < 8; j++)
    Out[(size_t)(rowBase + j) * FDIM + col] = fmaxf(c[j] + bn, 0.f);
}

// ---------- edge attention kernels ----------

// logits[e] = dot(q[dst], k[src]) / sqrt(128); atomic segment-max into mU[dst]
__global__ void __launch_bounds__(256)
k_edge_logits(const _Float16* __restrict__ Qh, const _Float16* __restrict__ Kh,
              const int* __restrict__ srcI, const int* __restrict__ dstI,
              float* __restrict__ logits, unsigned int* __restrict__ mU, int E)
{
  int e = blockIdx.x * 8 + (threadIdx.x >> 5);
  int lane = threadIdx.x & 31;
  if (e >= E) return;
  int s = srcI[e], d = dstI[e];
  v4h qv = *(const v4h*)(Qh + (size_t)d * FDIM + lane * 4);
  v4h kv = *(const v4h*)(Kh + (size_t)s * FDIM + lane * 4);
  float acc = (float)qv[0] * (float)kv[0] + (float)qv[1] * (float)kv[1]
            + (float)qv[2] * (float)kv[2] + (float)qv[3] * (float)kv[3];
#pragma unroll
  for (int o = 16; o > 0; o >>= 1) acc += __shfl_xor(acc, o, 32);
  if (lane == 0) {
    float lg = acc * 0.088388347648318447f;  // 1/sqrt(128)
    logits[e] = lg;
    atomicMax(mU + d, fkey(lg));
  }
}

// e = exp(logit - m[dst]); logits <- e; segment-sum into sSum[dst]
__global__ void k_edge_exp(const int* __restrict__ dstI, const unsigned int* __restrict__ mU,
                           float* __restrict__ logits, float* __restrict__ sSum, int E)
{
  int e = blockIdx.x * blockDim.x + threadIdx.x;
  if (e >= E) return;
  int d = dstI[e];
  float ex = __expf(logits[e] - funkey(mU[d]));
  logits[e] = ex;
  atomAddF(sSum + d, ex);
}

// AGG[dst] += (e / s[dst]) * v[src]   (AGG pre-initialized with the skip projection)
__global__ void __launch_bounds__(256)
k_edge_scatter(const _Float16* __restrict__ Vh,
               const int* __restrict__ srcI, const int* __restrict__ dstI,
               const float* __restrict__ eVal, const float* __restrict__ sSum,
               float* __restrict__ AGG, int E)
{
  int e = blockIdx.x * 8 + (threadIdx.x >> 5);
  int lane = threadIdx.x & 31;
  if (e >= E) return;
  int s = srcI[e], d = dstI[e];
  float alpha = eVal[e] / sSum[d];
  v4h vv = *(const v4h*)(Vh + (size_t)s * FDIM + lane * 4);
  float* row = AGG + (size_t)d * FDIM + lane * 4;
  atomAddF(row + 0, alpha * (float)vv[0]);
  atomAddF(row + 1, alpha * (float)vv[1]);
  atomAddF(row + 2, alpha * (float)vv[2]);
  atomAddF(row + 3, alpha * (float)vv[3]);
}

// h = relu(agg), stored as f16 for the next WMMA layer / pooling
__global__ void k_relu_f16(const float* __restrict__ in, _Float16* __restrict__ out, int n) {
  int i = blockIdx.x * blockDim.x + threadIdx.x;
  if (i < n) out[i] = (_Float16)fmaxf(in[i], 0.f);
}

// ---------- global attention pooling ----------

// gate[i] = dot(gate1[i], gw2) + gb2; atomic segment-max over graphs
__global__ void __launch_bounds__(256)
k_gate_dot(const float* __restrict__ G1, const float* __restrict__ gw2,
           const float* __restrict__ gb2, const int* __restrict__ batch,
           float* __restrict__ gate, unsigned int* __restrict__ gmaxU, int n)
{
  int i = blockIdx.x * 8 + (threadIdx.x >> 5);
  int lane = threadIdx.x & 31;
  if (i >= n) return;
  float4 h = *(const float4*)(G1 + (size_t)i * FDIM + lane * 4);
  float4 w = *(const float4*)(gw2 + lane * 4);
  float acc = h.x * w.x + h.y * w.y + h.z * w.z + h.w * w.w;
#pragma unroll
  for (int o = 16; o > 0; o >>= 1) acc += __shfl_xor(acc, o, 32);
  if (lane == 0) {
    float g = acc + gb2[0];
    gate[i] = g;
    atomicMax(gmaxU + batch[i], fkey(g));
  }
}

__global__ void k_gate_exp(const int* __restrict__ batch, const unsigned int* __restrict__ gmaxU,
                           float* __restrict__ gate, float* __restrict__ gsum, int n)
{
  int i = blockIdx.x * blockDim.x + threadIdx.x;
  if (i >= n) return;
  int b = batch[i];
  float ex = __expf(gate[i] - funkey(gmaxU[b]));
  gate[i] = ex;
  atomAddF(gsum + b, ex);
}

__global__ void __launch_bounds__(256)
k_pool(const _Float16* __restrict__ Hh, const int* __restrict__ batch,
       const float* __restrict__ gate, const float* __restrict__ gsum,
       float* __restrict__ pooled, int n)
{
  int i = blockIdx.x * 8 + (threadIdx.x >> 5);
  int lane = threadIdx.x & 31;
  if (i >= n) return;
  int b = batch[i];
  float w = gate[i] / gsum[b];
  v4h h = *(const v4h*)(Hh + (size_t)i * FDIM + lane * 4);
  float* row = pooled + (size_t)b * FDIM + lane * 4;
  atomAddF(row + 0, w * (float)h[0]);
  atomAddF(row + 1, w * (float)h[1]);
  atomAddF(row + 2, w * (float)h[2]);
  atomAddF(row + 3, w * (float)h[3]);
}

// out[g,c] = dot(pooled[g], cw[c]) + cb[c]   (64 x 10, tiny)
__global__ void k_cls(const float* __restrict__ pooled, const float* __restrict__ cw,
                      const float* __restrict__ cb, float* __restrict__ out)
{
  int t = blockIdx.x * blockDim.x + threadIdx.x;
  if (t >= NGRAPH * NCLS) return;
  int g = t / NCLS, c = t - g * NCLS;
  const float* pr = pooled + (size_t)g * FDIM;
  const float* wr = cw + (size_t)c * FDIM;
  float acc = cb[c];
#pragma unroll 16
  for (int k = 0; k < FDIM; k++) acc += pr[k] * wr[k];
  out[t] = acc;
}

// ---------- host orchestration ----------

static inline size_t al256(size_t x) { return (x + 255) & ~(size_t)255; }

extern "C" void kernel_launch(void* const* d_in, const int* in_sizes, int n_in,
                              void* d_out, int out_size, void* d_ws, size_t ws_size,
                              hipStream_t stream) {
  (void)in_sizes; (void)n_in; (void)out_size; (void)ws_size;

  const float* x     = (const float*)d_in[0];
  const int*   eidx  = (const int*)d_in[1];
  const int*   batch = (const int*)d_in[2];
  const float* w1[4] = {(const float*)d_in[3], (const float*)d_in[5], (const float*)d_in[7], (const float*)d_in[9]};
  const float* b1[4] = {(const float*)d_in[4], (const float*)d_in[6], (const float*)d_in[8], (const float*)d_in[10]};
  const float* w2[4] = {(const float*)d_in[11], (const float*)d_in[13], (const float*)d_in[15], (const float*)d_in[17]};
  const float* b2[4] = {(const float*)d_in[12], (const float*)d_in[14], (const float*)d_in[16], (const float*)d_in[18]};
  const float* gw1 = (const float*)d_in[19];
  const float* gb1 = (const float*)d_in[20];
  const float* gw2 = (const float*)d_in[21];
  const float* gb2 = (const float*)d_in[22];
  const float* cw  = (const float*)d_in[23];
  const float* cb  = (const float*)d_in[24];
  float* out = (float*)d_out;

  const int* srcI = eidx;            // edge_index[0]
  const int* dstI = eidx + N_EDGES;  // edge_index[1]

  // workspace carve-out (256B aligned)
  char* ws = (char*)d_ws;
  size_t off = 0;
  auto take = [&](size_t bytes) -> char* { char* p = ws + off; off = al256(off + bytes); return p; };
  _Float16* Xh   = (_Float16*)take((size_t)N_NODES * FDIM * 2);   // layer input (f16), reused as Hh
  _Float16* Qh   = (_Float16*)take((size_t)N_NODES * FDIM * 2);
  _Float16* Kh   = (_Float16*)take((size_t)N_NODES * FDIM * 2);
  _Float16* Vh   = (_Float16*)take((size_t)N_NODES * FDIM * 2);
  float*    AGG  = (float*)   take((size_t)N_NODES * FDIM * 4);   // skip-proj + attention agg (also gate1)
  float*    logi = (float*)   take((size_t)N_EDGES * 4);
  unsigned* mU   = (unsigned*)take((size_t)N_NODES * 4);
  float*    sSum = (float*)   take((size_t)N_NODES * 4);
  float*    gate = (float*)   take((size_t)N_NODES * 4);
  unsigned* gmaxU= (unsigned*)take((size_t)NGRAPH * 4);           // gmax|gsum|pooled contiguous
  float*    gsum = (float*)   take((size_t)NGRAPH * 4);
  float*    pool = (float*)   take((size_t)NGRAPH * FDIM * 4);
  _Float16* W1h  = (_Float16*)take((size_t)4 * FDIM * FDIM * 2);  // [512][128]
  float*    B1c  = (float*)   take((size_t)4 * FDIM * 4);         // [512]
  _Float16* W2h  = (_Float16*)take((size_t)4 * FDIM * FDIM * 2);
  float*    B2c  = (float*)   take((size_t)4 * FDIM * 4);
  _Float16* GW1h = (_Float16*)take((size_t)FDIM * FDIM * 2);

  const int WEL = FDIM * FDIM;                 // 16384 elems per weight
  const int NF  = N_NODES * FDIM;              // 12.8M
  // 1) convert weights + concat biases, convert X to f16
  for (int i = 0; i < 4; i++) {
    k_cvt_f16<<<(WEL + 255) / 256, 256, 0, stream>>>(w1[i], W1h + (size_t)i * WEL, WEL);
    k_copy_f32<<<1, FDIM, 0, stream>>>(b1[i], B1c + i * FDIM, FDIM);
    k_cvt_f16<<<(WEL + 255) / 256, 256, 0, stream>>>(w2[i], W2h + (size_t)i * WEL, WEL);
    k_copy_f32<<<1, FDIM, 0, stream>>>(b2[i], B2c + i * FDIM, FDIM);
  }
  k_cvt_f16<<<(WEL + 255) / 256, 256, 0, stream>>>(gw1, GW1h, WEL);
  k_cvt_f16<<<(NF + 255) / 256, 256, 0, stream>>>(x, Xh, NF);

  const int gemmBlocks = (N_NODES / 16) * 32 / 8;       // 25000 blocks of 8 waves
  const int edgeWBlocks = (N_EDGES + 7) / 8;
  const int nodeWBlocks = (N_NODES + 7) / 8;

  for (int layer = 0; layer < 2; layer++) {
    // reset per-node softmax state (key(0)=lower bound for any real float)
    k_fill_u32<<<(N_NODES + 255) / 256, 256, 0, stream>>>(mU, 0u, N_NODES);
    k_fill_u32<<<(N_NODES + 255) / 256, 256, 0, stream>>>((unsigned*)sSum, 0u, N_NODES);
    // fused Q|K|V|S projection (WMMA); S lands in AGG
    k_gemm_proj<<<gemmBlocks, 256, 0, stream>>>(Xh, layer ? W2h : W1h, layer ? B2c : B1c,
                                                Qh, Kh, Vh, AGG, N_NODES);
    // edge attention
    k_edge_logits<<<edgeWBlocks, 256, 0, stream>>>(Qh, Kh, srcI, dstI, logi, mU, N_EDGES);
    k_edge_exp<<<(N_EDGES + 255) / 256, 256, 0, stream>>>(dstI, mU, logi, sSum, N_EDGES);
    k_edge_scatter<<<edgeWBlocks, 256, 0, stream>>>(Vh, srcI, dstI, logi, sSum, AGG, N_EDGES);
    // h = relu(agg + skip), back to f16 input buffer
    k_relu_f16<<<(NF + 255) / 256, 256, 0, stream>>>(AGG, Xh, NF);
  }

  // global attention pooling (Xh now holds h2 in f16)
  k_gemm_relu<<<(N_NODES / 16) * 8 / 8, 256, 0, stream>>>(Xh, GW1h, gb1, AGG, N_NODES);
  k_fill_u32<<<(2 * NGRAPH + NGRAPH * FDIM + 255) / 256, 256, 0, stream>>>(gmaxU, 0u,
                                                    2 * NGRAPH + NGRAPH * FDIM);
  k_gate_dot<<<nodeWBlocks, 256, 0, stream>>>(AGG, gw2, gb2, batch, gate, gmaxU, N_NODES);
  k_gate_exp<<<(N_NODES + 255) / 256, 256, 0, stream>>>(batch, gmaxU, gate, gsum, N_NODES);
  k_pool<<<nodeWBlocks, 256, 0, stream>>>(Xh, batch, gate, gsum, pool, N_NODES);

  // classifier
  k_cls<<<(NGRAPH * NCLS + 255) / 256, 256, 0, stream>>>(pool, cw, cb, out);
}
